// TwoStageController_77068893160233
// MI455X (gfx1250) — matrix operand order repaired and checked by reading, not compiled, not run on hardware
//
#include <hip/hip_runtime.h>
#include <hip/hip_bf16.h>

typedef _Float16 h16_t;
typedef __attribute__((ext_vector_type(16))) _Float16 v16h;
typedef __attribute__((ext_vector_type(8)))  float    v8f;

#define DEVINL __device__ __forceinline__

DEVINL int lane_id() { return threadIdx.x & 31; }

DEVINL float wsum32(float v) {
#pragma unroll
  for (int m = 16; m >= 1; m >>= 1) v += __shfl_xor(v, m, 32);
  return v;
}

// A fragment: 16x32 f16 tile, A row-major with leading dim lda (K contiguous).
// lane<16: M=lane, K in {0..7,16..23}; lane>=16: M=lane-16, K in {8..15,24..31}
DEVINL v16h load_a(const h16_t* A, int lda) {
  int l = lane_id();
  const h16_t* row = A + (l & 15) * lda + ((l < 16) ? 0 : 8);
  v16h a;
#pragma unroll
  for (int j = 0; j < 4; ++j) { a[2 * j] = row[2 * j]; a[2 * j + 1] = row[2 * j + 1]; }
#pragma unroll
  for (int j = 0; j < 4; ++j) { a[8 + 2 * j] = row[16 + 2 * j]; a[8 + 2 * j + 1] = row[16 + 2 * j + 1]; }
  return a;
}

// B fragment: 32x16 tile given B-transpose rows (BT[n][k], k contiguous, ld=ldw).
// lane&15 = N, lanes<16: K in {0..7,16..23}, lanes>=16: K in {8..15,24..31}
DEVINL v16h load_bt(const h16_t* BT, int ldw) {
  int l = lane_id();
  const h16_t* row = BT + (l & 15) * ldw + ((l < 16) ? 0 : 8);
  v16h b;
#pragma unroll
  for (int j = 0; j < 4; ++j) { b[2 * j] = row[2 * j]; b[2 * j + 1] = row[2 * j + 1]; }
#pragma unroll
  for (int j = 0; j < 4; ++j) { b[8 + 2 * j] = row[16 + 2 * j]; b[8 + 2 * j + 1] = row[16 + 2 * j + 1]; }
  return b;
}

DEVINL v8f wmma_f16(v16h a, v16h b, v8f c) {
  return __builtin_amdgcn_wmma_f32_16x16x32_f16(false, a, false, b, (short)0, c, false, false);
}

// low 32 bits of a flat LDS pointer = LDS byte offset (aperture maps addr[31:0])
DEVINL unsigned lds_lo(const void* p) { return (unsigned)(unsigned long long)p; }

// CDNA5 async global->LDS copy (ASYNCcnt-tracked)
DEVINL void async_ld_b64(unsigned lds_off, const void* g) {
  asm volatile("global_load_async_to_lds_b64 %0, %1, off" :: "v"(lds_off), "v"(g) : "memory");
}
DEVINL void wait_async0() { asm volatile("s_wait_asynccnt 0" ::: "memory"); }

// ---------------- weight / embed conversion ----------------
__global__ void k_f32_to_f16(const float* __restrict__ s, h16_t* __restrict__ d, int n) {
  int i = blockIdx.x * blockDim.x + threadIdx.x;
  if (i < n) d[i] = (h16_t)s[i];
}

__global__ void k_embed(const int* __restrict__ seq, const float* __restrict__ embed,
                        h16_t* __restrict__ h16, int rows) {
  int i = blockIdx.x * blockDim.x + threadIdx.x;
  if (i >= rows * 64) return;
  int r = i >> 6, c = i & 63;
  h16[i] = (h16_t)embed[seq[r] * 64 + c];
}

// ---------------- WMMA GEMM: C = A[M,K] * W[N,K]^T + bias ----------------
// One wave computes a 64x16 strip (4 M-tiles) so B fragments are reused 4x.
// NK = K/32 compile-time so bf[] stays in fixed VGPRs (no movrels).
// FLAGS: bit0 = relu, bit1 = store f16 (else f32)
template <int NK, int FLAGS>
__global__ void k_gemm(const h16_t* __restrict__ A, int lda,
                       const h16_t* __restrict__ W, int ldw,
                       const float* __restrict__ bias,
                       void* __restrict__ C, int ldc,
                       int M, int N) {
  int gw = (int)((blockIdx.x * blockDim.x + threadIdx.x) >> 5);
  int tn_cnt = N >> 4;
  int strips = (M >> 6) * tn_cnt;
  if (gw >= strips) return;
  int tm = gw / tn_cnt, tn = gw % tn_cnt;
  const h16_t* Ab = A + (size_t)tm * 64 * lda;
  const h16_t* Wb = W + (size_t)tn * 16 * ldw;
  v16h bf[NK];
#pragma unroll
  for (int kk = 0; kk < NK; ++kk) bf[kk] = load_bt(Wb + kk * 32, ldw);
  v8f c[4] = {{}, {}, {}, {}};
#pragma unroll
  for (int m = 0; m < 4; ++m) {
#pragma unroll
    for (int kk = 0; kk < NK; ++kk)
      c[m] = wmma_f16(load_a(Ab + (size_t)m * 16 * lda + kk * 32, lda), bf[kk], c[m]);
  }
  int l = lane_id();
  int n = l & 15, mb = (l < 16) ? 0 : 8;
  float bv = bias ? bias[tn * 16 + n] : 0.0f;
#pragma unroll
  for (int m = 0; m < 4; ++m) {
#pragma unroll
    for (int j = 0; j < 8; ++j) {
      float v = c[m][j] + bv;
      if (FLAGS & 1) v = fmaxf(v, 0.0f);
      size_t idx = (size_t)(tm * 64 + m * 16 + mb + j) * ldc + tn * 16 + n;
      if (FLAGS & 2) ((h16_t*)C)[idx] = (h16_t)v;
      else           ((float*)C)[idx] = v;
    }
  }
}

// ---------------- flash attention per (batch, head) ----------------
// qkv layout: [B*L][192] f16, q at col 0, k at 64, v at 128; head offset 32*hd.
// K chunks staged via async load-to-LDS; V staged transposed. Raw f32 scores
// are bounced through LDS into the A-fragment layout so row max/sum become a
// local 16-wide reduction + ONE shfl_xor(16) (instead of 64 bpermutes/chunk),
// and exp(P) is produced directly in A-fragment order for the P@V WMMA.
__global__ void k_attn(const h16_t* __restrict__ qkv, h16_t* __restrict__ out) {
  __shared__ float sb_all[8][16 * 32]; // per-wave f32 score scratch (2KB each)
  __shared__ h16_t kch[32 * 32];       // K chunk, row-major [kv][dim]
  __shared__ h16_t vT[32 * 32];        // V chunk transposed [dim][kv]
  int tid = threadIdx.x;
  int wid = tid >> 5;
  int l = tid & 31;
  int b = blockIdx.x >> 1, hd = blockIdx.x & 1;
  const h16_t* qp = qkv + (size_t)b * 1024 * 192 + hd * 32;
  const h16_t* kp = qp + 64;
  const h16_t* vp = qp + 128;
  float* sb = sb_all[wid];
  int n = l & 15, mb = (l < 16) ? 0 : 8;
  int kb = (l < 16) ? 0 : 8; // A-layout K base for this lane
  v8f zc = {};
  // cooperative staging indices: 256 threads cover 32 rows x 8 4-half segments
  int srow = tid >> 3;          // 0..31 (kv row within chunk)
  int scol = (tid & 7) * 4;     // 0..28 (half offset within row)
  unsigned k_lds = lds_lo(&kch[srow * 32 + scol]);
  for (int i = 0; i < 8; ++i) {
    int qt = i * 8 + wid; // 64 q-tiles of 16 rows
    v16h qa = load_a(qp + (size_t)qt * 16 * 192, 192);
#pragma unroll
    for (int e = 0; e < 16; ++e) qa[e] = qa[e] * (h16_t)0.17677669529663687f; // fold 1/sqrt(32)
    float Mr = -1e30f, Lr = 0.0f; // running state for row (l&15), duplicated in both halves
    v8f o0 = {}, o1 = {};
    for (int ch = 0; ch < 32; ++ch) { // 32 kv-chunks of 32 rows
      // --- stage K (async copy) + V (transposed) into LDS ---
      const h16_t* gk = kp + ((size_t)ch * 32 + srow) * 192 + scol;
      async_ld_b64(k_lds, gk);
      const h16_t* gv = vp + ((size_t)ch * 32 + srow) * 192 + scol;
      h16_t v0 = gv[0], v1 = gv[1], v2 = gv[2], v3 = gv[3];
      vT[(scol + 0) * 32 + srow] = v0;
      vT[(scol + 1) * 32 + srow] = v1;
      vT[(scol + 2) * 32 + srow] = v2;
      vT[(scol + 3) * 32 + srow] = v3;
      wait_async0();
      __syncthreads();
      // --- scores (already scaled via Q) ---
      v8f s0 = wmma_f16(qa, load_bt(kch, 32), zc);            // kv rows +0..15
      v8f s1 = wmma_f16(qa, load_bt(kch + 16 * 32, 32), zc);  // kv rows +16..31
      // C layout -> LDS row-major [16][32] f32
#pragma unroll
      for (int j = 0; j < 8; ++j) {
        sb[(mb + j) * 32 + n] = s0[j];
        sb[(mb + j) * 32 + 16 + n] = s1[j];
      }
      // read back in A-fragment order: row n, K {kb..kb+7, 16+kb..16+kb+7}
      const float* srw = sb + n * 32 + kb;
      float sv[16];
#pragma unroll
      for (int jj = 0; jj < 8; ++jj) sv[jj] = srw[jj];
#pragma unroll
      for (int jj = 0; jj < 8; ++jj) sv[8 + jj] = srw[16 + jj];
      // row max: local 16 + partner half
      float cmax = sv[0];
#pragma unroll
      for (int jj = 1; jj < 16; ++jj) cmax = fmaxf(cmax, sv[jj]);
      cmax = fmaxf(cmax, __shfl_xor(cmax, 16, 32));
      float Mn = fmaxf(Mr, cmax);
      float alpha = __expf(Mr - Mn);
      float rsum = 0.0f;
      v16h pa;
#pragma unroll
      for (int jj = 0; jj < 16; ++jj) {
        float p = __expf(sv[jj] - Mn);
        rsum += p;
        pa[jj] = (h16_t)p;
      }
      rsum += __shfl_xor(rsum, 16, 32);
      Lr = Lr * alpha + rsum;
      Mr = Mn;
      // broadcast alpha (row state lives in lane mb+j) and rescale O
#pragma unroll
      for (int j = 0; j < 8; ++j) {
        float aj = __shfl(alpha, mb + j, 32);
        o0[j] *= aj; o1[j] *= aj;
      }
      o0 = wmma_f16(pa, load_bt(vT, 32), o0);            // head dims 0..15
      o1 = wmma_f16(pa, load_bt(vT + 16 * 32, 32), o1);  // head dims 16..31
      __syncthreads(); // protect kch/vT before next chunk staging
    }
    float inv = 1.0f / Lr;
    h16_t* ob = out + ((size_t)b * 1024 + (size_t)qt * 16) * 64 + hd * 32;
#pragma unroll
    for (int j = 0; j < 8; ++j) {
      float ij = __shfl(inv, mb + j, 32);
      ob[(size_t)(mb + j) * 64 + n] = (h16_t)(o0[j] * ij);
      ob[(size_t)(mb + j) * 64 + 16 + n] = (h16_t)(o1[j] * ij);
    }
  }
}

// ---------------- residual + LayerNorm (one wave per 64-wide row) ----------------
// mode 0: x = embed[seq[row]] + y ; mode 1: x = (float)hres[row] + y
__global__ void k_ln(const float* __restrict__ y,
                     const int* __restrict__ seq, const float* __restrict__ embed,
                     const h16_t* __restrict__ hres,
                     const float* __restrict__ g, const float* __restrict__ be,
                     h16_t* __restrict__ out16, float* __restrict__ out32,
                     int rows, int mode) {
  int row = blockIdx.x * 8 + (threadIdx.x >> 5);
  if (row >= rows) return;
  int l = lane_id();
  const float* yr = y + (size_t)row * 64;
  float x0, x1;
  if (mode == 0) {
    const float* er = embed + (size_t)seq[row] * 64;
    x0 = er[l] + yr[l]; x1 = er[l + 32] + yr[l + 32];
  } else {
    const h16_t* hr = hres + (size_t)row * 64;
    x0 = (float)hr[l] + yr[l]; x1 = (float)hr[l + 32] + yr[l + 32];
  }
  float m = wsum32(x0 + x1) * (1.0f / 64.0f);
  float d0 = x0 - m, d1 = x1 - m;
  float var = wsum32(d0 * d0 + d1 * d1) * (1.0f / 64.0f);
  float rs = rsqrtf(var + 1e-5f);
  float r0 = d0 * rs * g[l] + be[l];
  float r1 = d1 * rs * g[l + 32] + be[l + 32];
  if (out16) { h16_t* o = out16 + (size_t)row * 64; o[l] = (h16_t)r0; o[l + 32] = (h16_t)r1; }
  if (out32) { float* o = out32 + (size_t)row * 64; o[l] = r0; o[l + 32] = r1; }
}

// ---------------- gates + top-6 + memory reader + CE, one block per batch ----------------
__global__ void k_select_reader(const float* __restrict__ hidden,
                                const int* __restrict__ query, const int* __restrict__ target,
                                const float* __restrict__ w_s1, const float* __restrict__ b_s1,
                                const float* __restrict__ w_s2, const float* __restrict__ b_s2,
                                const float* __restrict__ w_qr, const float* __restrict__ b_qr,
                                const float* __restrict__ w_or, const float* __restrict__ b_or,
                                const float* __restrict__ q_embed,
                                float* __restrict__ loss_out) {
  __shared__ float s2f[1024];
  __shared__ float rv[256];
  __shared__ int ri[256];
  __shared__ int top[6];
  __shared__ float mem[6][64];
  __shared__ float qv[64];
  __shared__ float ctx[64];
  __shared__ float sw[6];
  __shared__ float lg[64];
  int b = blockIdx.x, t = threadIdx.x;
  const float* hb = hidden + (size_t)b * 1024 * 64;
  for (int i = t; i < 1024; i += 256) {
    const float* row = hb + (size_t)i * 64;
    float a1 = b_s1[0], a2 = b_s2[0];
    for (int k = 0; k < 64; ++k) { float x = row[k]; a1 += w_s1[k] * x; a2 += w_s2[k] * x; }
    float sg2 = 1.0f / (1.0f + __expf(-a2));
    s2f[i] = (a1 > 0.0f) ? sg2 : 0.0f; // sigmoid(a1)>0.5 <=> a1>0
  }
  __syncthreads();
  for (int s = 0; s < 6; ++s) { // iterative argmax, smallest index on ties (top_k order)
    float bv = -1e30f; int bi = 0x7fffffff;
    for (int i = t; i < 1024; i += 256) {
      float v = s2f[i];
      if (v > bv || (v == bv && i < bi)) { bv = v; bi = i; }
    }
    rv[t] = bv; ri[t] = bi;
    __syncthreads();
    for (int off = 128; off >= 1; off >>= 1) {
      if (t < off) {
        float ov = rv[t + off]; int oi = ri[t + off];
        if (ov > rv[t] || (ov == rv[t] && oi < ri[t])) { rv[t] = ov; ri[t] = oi; }
      }
      __syncthreads();
    }
    if (t == 0) { top[s] = ri[0]; s2f[ri[0]] = -2e30f; }
    __syncthreads();
  }
  if (t < 64) {
    for (int s = 0; s < 6; ++s) mem[s][t] = hb[(size_t)top[s] * 64 + t];
    const float* qe = q_embed + (size_t)query[b] * 64;
    float acc = b_qr[t];
    for (int k = 0; k < 64; ++k) acc += w_qr[t * 64 + k] * qe[k];
    qv[t] = acc;
  }
  __syncthreads();
  if (t < 6) {
    float s = 0.0f;
    for (int k = 0; k < 64; ++k) s += qv[k] * mem[t][k];
    sw[t] = s * 0.125f; // 1/sqrt(64)
  }
  __syncthreads();
  if (t == 0) {
    float mx = sw[0];
    for (int s = 1; s < 6; ++s) mx = fmaxf(mx, sw[s]);
    float den = 0.0f;
    for (int s = 0; s < 6; ++s) { sw[s] = __expf(sw[s] - mx); den += sw[s]; }
    float inv = 1.0f / den;
    for (int s = 0; s < 6; ++s) sw[s] *= inv;
  }
  __syncthreads();
  if (t < 64) {
    float c = 0.0f;
    for (int s = 0; s < 6; ++s) c += sw[s] * mem[s][t];
    ctx[t] = c;
  }
  __syncthreads();
  if (t < 64) {
    float a = b_or[t]; const float* wr = w_or + (size_t)t * 64;
    for (int k = 0; k < 64; ++k) a += wr[k] * ctx[k];
    lg[t] = a;
  }
  __syncthreads();
  if (t == 0) {
    float mx = lg[0];
    for (int v = 1; v < 64; ++v) mx = fmaxf(mx, lg[v]);
    float den = 0.0f;
    for (int v = 0; v < 64; ++v) den += __expf(lg[v] - mx);
    loss_out[b] = -(lg[target[b]] - mx - __logf(den));
  }
}

__global__ void k_mean64(const float* __restrict__ loss, float* __restrict__ out) {
  __shared__ float sh[64];
  sh[threadIdx.x] = loss[threadIdx.x];
  __syncthreads();
  if (threadIdx.x == 0) {
    float s = 0.0f;
    for (int i = 0; i < 64; ++i) s += sh[i];
    *out = s / 64.0f;
  }
}

extern "C" void kernel_launch(void* const* d_in, const int* in_sizes, int n_in,
                              void* d_out, int out_size, void* d_ws, size_t ws_size,
                              hipStream_t stream) {
  const int ROWS = 64 * 1024; // B*L
  const int* seq = (const int*)d_in[0];
  const int* query = (const int*)d_in[1];
  const int* target = (const int*)d_in[2];
  const float* embed = (const float*)d_in[3];
  const float* wqkv = (const float*)d_in[4];
  const float* bqkv = (const float*)d_in[5];
  const float* wo = (const float*)d_in[6];
  const float* bo = (const float*)d_in[7];
  const float* wff1 = (const float*)d_in[8];
  const float* bff1 = (const float*)d_in[9];
  const float* wff2 = (const float*)d_in[10];
  const float* bff2 = (const float*)d_in[11];
  const float* g1 = (const float*)d_in[12];
  const float* be1 = (const float*)d_in[13];
  const float* g2 = (const float*)d_in[14];
  const float* be2 = (const float*)d_in[15];
  const float* w_s1 = (const float*)d_in[16];
  const float* b_s1 = (const float*)d_in[17];
  const float* w_s2 = (const float*)d_in[18];
  const float* b_s2 = (const float*)d_in[19];
  const float* w_qr = (const float*)d_in[20];
  const float* b_qr = (const float*)d_in[21];
  const float* w_or = (const float*)d_in[22];
  const float* b_or = (const float*)d_in[23];
  const float* q_embed = (const float*)d_in[24];
  (void)in_sizes; (void)n_in; (void)out_size; (void)ws_size;

  char* ws = (char*)d_ws;
  size_t off = 0;
  auto alloc = [&](size_t bytes) -> char* {
    char* p = ws + off;
    off = (off + bytes + 255) & ~(size_t)255;
    return p;
  };
  h16_t* h16 = (h16_t*)alloc((size_t)ROWS * 64 * 2);
  h16_t* qkv16 = (h16_t*)alloc((size_t)ROWS * 192 * 2);
  h16_t* ao16 = (h16_t*)alloc((size_t)ROWS * 64 * 2);
  h16_t* h1_16 = (h16_t*)alloc((size_t)ROWS * 64 * 2);
  h16_t* a1_16 = (h16_t*)alloc((size_t)ROWS * 128 * 2);
  float* y32 = (float*)alloc((size_t)ROWS * 64 * 4);
  float* hid32 = (float*)alloc((size_t)ROWS * 64 * 4);
  h16_t* wqkv16 = (h16_t*)alloc(192 * 64 * 2);
  h16_t* wo16 = (h16_t*)alloc(64 * 64 * 2);
  h16_t* wff1_16 = (h16_t*)alloc(128 * 64 * 2);
  h16_t* wff2_16 = (h16_t*)alloc(64 * 128 * 2);
  float* lossb = (float*)alloc(64 * 4);

  k_f32_to_f16<<<(192 * 64 + 255) / 256, 256, 0, stream>>>(wqkv, wqkv16, 192 * 64);
  k_f32_to_f16<<<(64 * 64 + 255) / 256, 256, 0, stream>>>(wo, wo16, 64 * 64);
  k_f32_to_f16<<<(128 * 64 + 255) / 256, 256, 0, stream>>>(wff1, wff1_16, 128 * 64);
  k_f32_to_f16<<<(64 * 128 + 255) / 256, 256, 0, stream>>>(wff2, wff2_16, 64 * 128);

  k_embed<<<(ROWS * 64 + 255) / 256, 256, 0, stream>>>(seq, embed, h16, ROWS);

  { // qkv = h @ wqkv^T + bqkv  -> f16 [ROWS][192]
    int strips = (ROWS / 64) * (192 / 16);
    k_gemm<2, 2><<<(strips + 7) / 8, 256, 0, stream>>>(h16, 64, wqkv16, 64, bqkv, qkv16, 192, ROWS, 192);
  }
  k_attn<<<128, 256, 0, stream>>>(qkv16, ao16);
  { // y = attn_o @ wo^T + bo -> f32
    int strips = (ROWS / 64) * (64 / 16);
    k_gemm<2, 0><<<(strips + 7) / 8, 256, 0, stream>>>(ao16, 64, wo16, 64, bo, y32, 64, ROWS, 64);
  }
  k_ln<<<ROWS / 8, 256, 0, stream>>>(y32, seq, embed, nullptr, g1, be1, h1_16, nullptr, ROWS, 0);
  { // ff1 = relu(h1 @ w_ff1^T + b) -> f16 [ROWS][128]
    int strips = (ROWS / 64) * (128 / 16);
    k_gemm<2, 3><<<(strips + 7) / 8, 256, 0, stream>>>(h1_16, 64, wff1_16, 64, bff1, a1_16, 128, ROWS, 128);
  }
  { // y = ff1 @ w_ff2^T + b -> f32 (K=128)
    int strips = (ROWS / 64) * (64 / 16);
    k_gemm<4, 0><<<(strips + 7) / 8, 256, 0, stream>>>(a1_16, 128, wff2_16, 128, bff2, y32, 64, ROWS, 64);
  }
  k_ln<<<ROWS / 8, 256, 0, stream>>>(y32, nullptr, nullptr, h1_16, g2, be2, nullptr, hid32, ROWS, 1);

  k_select_reader<<<64, 256, 0, stream>>>(hid32, query, target, w_s1, b_s1, w_s2, b_s2,
                                          w_qr, b_qr, w_or, b_or, q_embed, lossb);
  k_mean64<<<1, 64, 0, stream>>>(lossb, (float*)d_out);
}
